// Gemma2QuantizedAttention_71700184040136
// MI455X (gfx1250) — matrix operand order repaired and checked by reading, not compiled, not run on hardware
//
#include <hip/hip_runtime.h>
#include <hip/hip_bf16.h>
#include <math.h>

typedef __attribute__((ext_vector_type(16))) __bf16 v16bf;
typedef __attribute__((ext_vector_type(8)))  float  v8f;
typedef __attribute__((address_space(3))) char lds_char;

#define B_ 2
#define S_ 2048
#define H_ 3584
#define NH_ 16
#define KVH_ 8
#define D_ 256
#define SCALE_ 0.0625f      /* 256^-0.5 */
#define SOFTCAP_ 50.0f
#define NEG_ (-1.0e9f)

union U16BF { uint4 q[2]; v16bf v; };
union U8BF  { uint4 q; __bf16 h[8]; };

__device__ __forceinline__ v8f wmma_bf16(v16bf a, v16bf b, v8f c) {
    return __builtin_amdgcn_wmma_f32_16x16x32_bf16(false, a, false, b, (short)0, c, false, false);
}

// gfx1250 async copy global -> LDS (16B per lane), tracked by ASYNCcnt
__device__ __forceinline__ void async_b128_to_lds(void* lds, const void* glb) {
    unsigned dst = (unsigned)(unsigned long long)(lds_char*)lds;
    asm volatile("global_load_async_to_lds_b128 %0, %1, off"
                 :: "v"(dst), "v"((unsigned long long)glb) : "memory");
}
__device__ __forceinline__ void wait_async0() {
    asm volatile("s_wait_asynccnt 0" ::: "memory");
}

// ---------------------------------------------------------------- convert ---
__global__ void cvt_f32_bf16(const float* __restrict__ in, __bf16* __restrict__ out, int n4) {
    int i = blockIdx.x * blockDim.x + threadIdx.x;
    if (i >= n4) return;
    float4 f = ((const float4*)in)[i];
    __bf16* o = out + (long)i * 4;
    o[0] = (__bf16)f.x; o[1] = (__bf16)f.y; o[2] = (__bf16)f.z; o[3] = (__bf16)f.w;
}

// ------------------------------------------------------------------- rope ---
// x layout: [b][head][s][d], D=256, rotate-half pairs (j, j+128)
__global__ void rope_bf16(__bf16* __restrict__ x, long total) {
    long idx = (long)blockIdx.x * blockDim.x + threadIdx.x;
    if (idx >= total) return;
    int j = (int)(idx & 127);
    int s = (int)((idx >> 7) & (S_ - 1));
    long hb = idx >> 18;                       // 7 bits j + 11 bits s
    __bf16* row = x + (hb * S_ + s) * D_;
    float inv = __expf(-(float)j * (logf(10000.0f) / 128.0f));
    float ang = (float)s * inv;
    float c = cosf(ang), sn = sinf(ang);
    float x1 = (float)row[j], x2 = (float)row[j + 128];
    row[j]       = (__bf16)(x1 * c - x2 * sn);
    row[j + 128] = (__bf16)(x2 * c + x1 * sn);
}

// ------------------------------------------------------------------- gemm ---
// C[m,n] = sum_k A[m,k] * W[n,k]; A,W bf16 row-major.
// Wave tile 32x64 (8 accumulators), WG tile 256x64, double-buffered async B.
// headMode: write bf16 to layout ((n>>8)*S + m)*256 + (n&255); else fp32.
__global__ __launch_bounds__(256) void gemm_bf16(
    const __bf16* __restrict__ A, long aBatch, int lda,
    const __bf16* __restrict__ W, int ldw, int K,
    float* __restrict__ outF, __bf16* __restrict__ outB,
    int headMode, long outBatch, int ldout)
{
    __shared__ __bf16 Bt[2][64 * 32];
    const int tid  = threadIdx.x;
    const int w    = tid >> 5;
    const int lane = tid & 31;
    const int lm   = lane & 15;
    const int hi   = (lane & 16) ? 1 : 0;
    const __bf16* Ab = A + (long)blockIdx.z * aBatch;
    const int m0 = blockIdx.x * 256 + w * 32;
    const int n0 = blockIdx.y * 64;
    const int mrow0 = m0 + lm;
    const int mrow1 = m0 + 16 + lm;
    const int khalf = hi ? 8 : 0;

    v8f acc0[4], acc1[4];
#pragma unroll
    for (int t = 0; t < 4; t++) {
        acc0[t] = (v8f){0.f,0.f,0.f,0.f,0.f,0.f,0.f,0.f};
        acc1[t] = (v8f){0.f,0.f,0.f,0.f,0.f,0.f,0.f,0.f};
    }

    const int sn   = tid >> 2;            // staging row 0..63
    const int skof = (tid & 3) * 8;       // staging k-offset

    auto stageB = [&](int buf, int kk) {
        async_b128_to_lds(&Bt[buf][sn * 32 + skof],
                          W + (long)(n0 + sn) * ldw + kk + skof);
    };

    stageB(0, 0);
    wait_async0();
    __syncthreads();

    int cur = 0;
    for (int k0 = 0; k0 < K; k0 += 32) {
        if (k0 + 32 < K) stageB(cur ^ 1, k0 + 32);   // overlap with compute

        U16BF a0, a1;
        const __bf16* ap0 = Ab + (long)mrow0 * lda + k0 + khalf;
        const __bf16* ap1 = Ab + (long)mrow1 * lda + k0 + khalf;
        __builtin_prefetch(ap0 + 32, 0, 1);
        a0.q[0] = *(const uint4*)ap0; a0.q[1] = *(const uint4*)(ap0 + 16);
        a1.q[0] = *(const uint4*)ap1; a1.q[1] = *(const uint4*)(ap1 + 16);

#pragma unroll
        for (int t = 0; t < 4; t++) {
            U16BF bb;  // WMMA B-layout from LDS
            const __bf16* bp = &Bt[cur][(t * 16 + lm) * 32 + (hi ? 8 : 0)];
            bb.q[0] = *(const uint4*)bp;
            bb.q[1] = *(const uint4*)(bp + 16);
            acc0[t] = wmma_bf16(a0.v, bb.v, acc0[t]);
            acc1[t] = wmma_bf16(a1.v, bb.v, acc1[t]);
        }
        wait_async0();
        __syncthreads();
        cur ^= 1;
    }

    long zb = (long)blockIdx.z * outBatch;
#pragma unroll
    for (int t = 0; t < 4; t++) {
#pragma unroll
        for (int r = 0; r < 8; r++) {
            int n = n0 + t * 16 + lm;
            int ma = m0 + r + hi * 8;
            int mb = ma + 16;
            float va = acc0[t][r], vb = acc1[t][r];
            if (headMode) {
                long base = zb + (long)(n >> 8) * S_ * 256 + (n & 255);
                outB[base + (long)ma * 256] = (__bf16)va;
                outB[base + (long)mb * 256] = (__bf16)vb;
            } else {
                outF[zb + (long)ma * ldout + n] = va;
                outF[zb + (long)mb * ldout + n] = vb;
            }
        }
    }
}

// -------------------------------------------------------------- attention ---
// Q/K/V bf16 [b][head][s][256] (post-rope). O bf16 [b][s][NH*256].
__global__ __launch_bounds__(128) void attn_kernel(
    const __bf16* __restrict__ Qg, const __bf16* __restrict__ Kg,
    const __bf16* __restrict__ Vg, __bf16* __restrict__ Og)
{
    __shared__ __bf16 Kt[32 * 256];    // [key][d] row-major
    __shared__ __bf16 Vt[256 * 32];    // [d][key] (transposed)
    __shared__ __bf16 Pl[4 * 16 * 32]; // per-wave P staging

    const int tid  = threadIdx.x;
    const int w    = tid >> 5;
    const int lane = tid & 31;
    const int lm   = lane & 15;
    const int hi   = (lane & 16) ? 1 : 0;
    const int q0   = blockIdx.x * 64;
    const int h    = blockIdx.y;
    const int b    = blockIdx.z;
    const int kvh  = h >> 1;                  // GROUPS = 2
    const int qm   = q0 + w * 16;
    const int kh   = hi ? 8 : 0;

    // Q resident in A-layout: 8 chunks of 16x32
    v16bf qa[8];
    {
        const __bf16* qrow = Qg + (((long)b * NH_ + h) * S_ + qm + lm) * D_;
#pragma unroll
        for (int c = 0; c < 8; c++) {
            U16BF u;
            u.q[0] = *(const uint4*)(qrow + c * 32 + kh);
            u.q[1] = *(const uint4*)(qrow + c * 32 + 16 + kh);
            qa[c] = u.v;
        }
    }

    float mrow[8], lrow[8];
    v8f Ot[16];
#pragma unroll
    for (int r = 0; r < 8; r++) { mrow[r] = -1e30f; lrow[r] = 0.f; }
#pragma unroll
    for (int t = 0; t < 16; t++) Ot[t] = (v8f){0.f,0.f,0.f,0.f,0.f,0.f,0.f,0.f};

    const __bf16* Kbase = Kg + ((long)b * KVH_ + kvh) * S_ * D_;
    const __bf16* Vbase = Vg + ((long)b * KVH_ + kvh) * S_ * D_;
    __bf16* Pw = &Pl[w * 16 * 32];

    const int kend = q0 + 64;                 // causal upper bound for this WG
    for (int k0 = 0; k0 < kend; k0 += 32) {
        { // cooperative stage: K async row-major, V manual transposed
            int key = tid >> 2, dbase = (tid & 3) * 64;
            const __bf16* krow = Kbase + (long)(k0 + key) * D_ + dbase;
            const __bf16* vrow = Vbase + (long)(k0 + key) * D_ + dbase;
#pragma unroll
            for (int i = 0; i < 8; i++)
                async_b128_to_lds(&Kt[key * 256 + dbase + i * 8], krow + i * 8);
#pragma unroll
            for (int i = 0; i < 8; i++) {
                U8BF vv; vv.q = *(const uint4*)(vrow + i * 8);
#pragma unroll
                for (int jj = 0; jj < 8; jj++)
                    Vt[(dbase + i * 8 + jj) * 32 + key] = vv.h[jj];
            }
            wait_async0();
        }
        __syncthreads();

        // scores: two 16x16 tiles over 32 keys, K-dim = d (256)
        v8f sc[2];
#pragma unroll
        for (int nt = 0; nt < 2; nt++) {
            v8f s = (v8f){0.f,0.f,0.f,0.f,0.f,0.f,0.f,0.f};
#pragma unroll
            for (int c = 0; c < 8; c++) {
                U16BF bb;
                const __bf16* bp = &Kt[(nt * 16 + lm) * 256 + c * 32 + (hi ? 8 : 0)];
                bb.q[0] = *(const uint4*)bp;
                bb.q[1] = *(const uint4*)(bp + 16);
                s = wmma_bf16(qa[c], bb.v, s);
            }
            sc[nt] = s;
        }

        // soft-cap, causal mask, online softmax (16-lane row reductions)
        float p0[8], p1[8];
#pragma unroll
        for (int r = 0; r < 8; r++) {
            int m_abs = qm + r + hi * 8;
            float x0 = SOFTCAP_ * tanhf(sc[0][r] * (SCALE_ / SOFTCAP_));
            float x1 = SOFTCAP_ * tanhf(sc[1][r] * (SCALE_ / SOFTCAP_));
            if (k0 + lm      > m_abs) x0 = NEG_;
            if (k0 + 16 + lm > m_abs) x1 = NEG_;
            float mx = fmaxf(x0, x1);
#pragma unroll
            for (int off = 8; off >= 1; off >>= 1)
                mx = fmaxf(mx, __shfl_xor(mx, off, 32));
            float mnew  = fmaxf(mrow[r], mx);
            float alpha = __expf(mrow[r] - mnew);
            float e0 = __expf(x0 - mnew), e1 = __expf(x1 - mnew);
            float sum = e0 + e1;
#pragma unroll
            for (int off = 8; off >= 1; off >>= 1)
                sum += __shfl_xor(sum, off, 32);
            lrow[r] = lrow[r] * alpha + sum;
            mrow[r] = mnew;
            p0[r] = e0; p1[r] = e1;
#pragma unroll
            for (int t = 0; t < 16; t++) Ot[t][r] *= alpha;
        }

        // reshape P: C-layout -> row-major LDS -> A-layout
#pragma unroll
        for (int r = 0; r < 8; r++) {
            int ml = r + hi * 8;
            Pw[ml * 32 + lm]      = (__bf16)p0[r];
            Pw[ml * 32 + 16 + lm] = (__bf16)p1[r];
        }
        __syncthreads();

        { // O += P * V, K-dim = key (32)
            U16BF pa;
            const __bf16* pp = &Pw[lm * 32 + kh];
            pa.q[0] = *(const uint4*)pp;
            pa.q[1] = *(const uint4*)(pp + 16);
#pragma unroll
            for (int t = 0; t < 16; t++) {
                U16BF vb;
                const __bf16* vp = &Vt[(t * 16 + lm) * 32 + (hi ? 8 : 0)];
                vb.q[0] = *(const uint4*)vp;
                vb.q[1] = *(const uint4*)(vp + 16);
                Ot[t] = wmma_bf16(pa.v, vb.v, Ot[t]);
            }
        }
        __syncthreads();
    }

    // normalize and store attn output bf16 [b][s][NH*256]
#pragma unroll
    for (int r = 0; r < 8; r++) {
        float inv = 1.0f / lrow[r];
        int srow = qm + r + hi * 8;
        __bf16* orow = Og + ((long)b * S_ + srow) * (NH_ * D_) + h * D_;
#pragma unroll
        for (int t = 0; t < 16; t++)
            orow[t * 16 + lm] = (__bf16)(Ot[t][r] * inv);
    }
}

// ----------------------------------------------------------------- launch ---
extern "C" void kernel_launch(void* const* d_in, const int* in_sizes, int n_in,
                              void* d_out, int out_size, void* d_ws, size_t ws_size,
                              hipStream_t stream) {
    (void)in_sizes; (void)n_in; (void)out_size; (void)ws_size;
    const float* hs = (const float*)d_in[0];
    const float* Wq = (const float*)d_in[1];
    const float* Wk = (const float*)d_in[2];
    const float* Wv = (const float*)d_in[3];
    const float* Wo = (const float*)d_in[4];
    float* out = (float*)d_out;

    const long nHS = (long)B_ * S_ * H_;           // 14,680,064
    const long nWq = (long)NH_ * D_ * H_;          // 14,680,064
    const long nWk = (long)KVH_ * D_ * H_;         //  7,340,032
    const long nWo = (long)H_ * NH_ * D_;          // 14,680,064
    const long nQ  = (long)B_ * NH_ * S_ * D_;     // 16,777,216
    const long nK  = (long)B_ * KVH_ * S_ * D_;    //  8,388,608
    const long nAt = (long)B_ * S_ * NH_ * D_;     // 16,777,216

    char* p = (char*)d_ws;
    __bf16* hsB = (__bf16*)p;              p += nHS * 2;
    __bf16* WqB = (__bf16*)p;              p += nWq * 2;
    __bf16* WkB = (__bf16*)p;              p += nWk * 2;
    __bf16* WvB = (__bf16*)p;              p += nWk * 2;
    __bf16* WoB = (__bf16*)p;              p += nWo * 2;
    __bf16* qB  = (__bf16*)p;              p += nQ  * 2;
    __bf16* kB  = (__bf16*)p;              p += nK  * 2;
    __bf16* vB  = (__bf16*)p;              p += nK  * 2;
    __bf16* atB = (__bf16*)p;              p += nAt * 2;

    auto cvt = [&](const float* src, __bf16* dst, long n) {
        int n4 = (int)(n / 4);
        cvt_f32_bf16<<<(n4 + 255) / 256, 256, 0, stream>>>(src, dst, n4);
    };
    cvt(hs, hsB, nHS);
    cvt(Wq, WqB, nWq);
    cvt(Wk, WkB, nWk);
    cvt(Wv, WvB, nWk);
    cvt(Wo, WoB, nWo);

    // Q projection: [S x H] * [4096 x H]^T -> per-head bf16
    gemm_bf16<<<dim3(S_ / 256, (NH_ * D_) / 64, B_), 256, 0, stream>>>(
        hsB, (long)S_ * H_, H_, WqB, H_, H_,
        nullptr, qB, 1, (long)NH_ * S_ * D_, 0);
    // K projection
    gemm_bf16<<<dim3(S_ / 256, (KVH_ * D_) / 64, B_), 256, 0, stream>>>(
        hsB, (long)S_ * H_, H_, WkB, H_, H_,
        nullptr, kB, 1, (long)KVH_ * S_ * D_, 0);
    // V projection
    gemm_bf16<<<dim3(S_ / 256, (KVH_ * D_) / 64, B_), 256, 0, stream>>>(
        hsB, (long)S_ * H_, H_, WvB, H_, H_,
        nullptr, vB, 1, (long)KVH_ * S_ * D_, 0);

    // RoPE in place on q and k
    {
        long tq = (long)B_ * NH_ * S_ * 128;
        rope_bf16<<<(unsigned)((tq + 255) / 256), 256, 0, stream>>>(qB, tq);
        long tk = (long)B_ * KVH_ * S_ * 128;
        rope_bf16<<<(unsigned)((tk + 255) / 256), 256, 0, stream>>>(kB, tk);
    }

    // flash attention
    attn_kernel<<<dim3(S_ / 64, NH_, B_), 128, 0, stream>>>(qB, kB, vB, atB);

    // output projection: [S x 4096] * [H x 4096]^T -> fp32 out
    gemm_bf16<<<dim3(S_ / 256, H_ / 64, B_), 256, 0, stream>>>(
        atB, (long)S_ * (NH_ * D_), NH_ * D_, WoB, NH_ * D_, NH_ * D_,
        out, nullptr, 0, (long)S_ * H_, H_);
}
